// MixtralMoeBlock_28286654611541
// MI455X (gfx1250) — matrix operand order repaired and checked by reading, not compiled, not run on hardware
//
#include <hip/hip_runtime.h>
#include <cstdint>
#include <cstddef>

// ---------------- problem constants ----------------
constexpr int BATCH = 4;
constexpr int SEQ   = 2048;
constexpr int HID   = 1024;
constexpr int FFN   = 3584;
constexpr int NEXP  = 8;
constexpr int TOPK  = 2;
constexpr int CAP   = 640;                 // int(SEQ*TOPK/NEXP * 1.25)
constexpr int TOKENS = BATCH * SEQ;        // 8192
constexpr int ROWS   = NEXP * BATCH * CAP; // 20480 capacity rows total

// ---------------- workspace layout (bytes) ----------------
constexpr size_t NW      = (size_t)NEXP * HID * FFN;            // elements per weight tensor
constexpr size_t OFF_W1  = 0;
constexpr size_t OFF_W3  = OFF_W1 + NW * 2;
constexpr size_t OFF_W2  = OFF_W3 + NW * 2;
constexpr size_t OFF_DSP = OFF_W2 + NW * 2;                     // u16  ROWS*HID
constexpr size_t OFF_ACT = OFF_DSP + (size_t)ROWS * HID * 2;    // u16  ROWS*FFN
constexpr size_t OFF_EO  = OFF_ACT + (size_t)ROWS * FFN * 2;    // f32  ROWS*HID
constexpr size_t OFF_TE  = OFF_EO  + (size_t)ROWS * HID * 4;    // i32  TOKENS*TOPK
constexpr size_t OFF_TP  = OFF_TE  + (size_t)TOKENS * TOPK * 4; // f32  TOKENS*TOPK
constexpr size_t OFF_SL  = OFF_TP  + (size_t)TOKENS * TOPK * 4; // i32  TOKENS*TOPK
constexpr size_t OFF_DT  = OFF_SL  + (size_t)TOKENS * TOPK * 4; // i32  ROWS

// ---------------- WMMA / TDM vector types (CDNA5 wave32) ----------------
typedef __attribute__((ext_vector_type(16))) __bf16 bf16x16;
typedef __attribute__((ext_vector_type(8)))  float  floatx8;
typedef __attribute__((ext_vector_type(4)))  unsigned int u32x4;
typedef __attribute__((ext_vector_type(8)))  int    i32x8;
typedef __attribute__((ext_vector_type(4)))  int    i32x4;

union Frag {
    bf16x16 v;
    uint4   q[2];
};

__device__ __forceinline__ unsigned short f2bf(float f) {
    unsigned int u = __float_as_uint(f);
    u += 0x7FFFu + ((u >> 16) & 1u);   // round-to-nearest-even
    return (unsigned short)(u >> 16);
}

// ---------------- Tensor Data Mover: 2D tile (bf16) global -> LDS ----------------
// Loads tile_rows x tile_k (16-bit elements) from `gaddr` (row stride
// `row_stride_elems`) into LDS at `lds_off`, inserting 16B of padding after
// every 64B row so the LDS image matches As[TM][TK+8].
__device__ __forceinline__ void tdm_load_tile_bf16(
        unsigned lds_off, const void* gaddr,
        int tile_k, int tile_rows, int row_stride_elems) {
    unsigned long long ga = (unsigned long long)(uintptr_t)gaddr;
    u32x4 g0;
    g0[0] = 1u;                                   // count=1, user descriptor
    g0[1] = lds_off;                              // lds_addr (bytes)
    g0[2] = (unsigned)ga;                         // global_addr[31:0]
    g0[3] = (unsigned)(ga >> 32) | (2u << 30);    // global_addr[56:32] | type=2
    i32x8 g1;
    g1[0] = (1 << 16)                             // data_size = 2 bytes
          | (1 << 20)                             // pad_enable
          | (3 << 22)                             // pad_interval: 16 DWORDs (64B)
          | (3 << 25);                            // pad_amount : 4 DWORDs (16B)
    g1[1] = (tile_k & 0xFFFF) << 16;              // tensor_dim0[15:0]  @ bits 63:48
    g1[2] = (tile_rows & 0xFFFF) << 16;           // tensor_dim1[15:0]  @ bits 95:80
    g1[3] = (tile_k & 0xFFFF) << 16;              // tile_dim0          @ bits 127:112
    g1[4] = (tile_rows & 0xFFFF);                 // tile_dim1          @ bits 143:128
    g1[5] = row_stride_elems;                     // tensor_dim0_stride[31:0]
    g1[6] = 0;
    g1[7] = 0;
    i32x4 z4 = {0, 0, 0, 0};
#if defined(__clang_major__) && (__clang_major__ >= 23)
    i32x8 z8 = {0, 0, 0, 0, 0, 0, 0, 0};
    __builtin_amdgcn_tensor_load_to_lds(g0, g1, z4, z4, z8, 0);
#else
    __builtin_amdgcn_tensor_load_to_lds(g0, g1, z4, z4, 0);
#endif
}

// ---------------- fp32 -> bf16 conversion ----------------
__global__ void cvt_bf16_kernel(const float* __restrict__ src,
                                unsigned short* __restrict__ dst, long n) {
    long i = ((long)blockIdx.x * blockDim.x + threadIdx.x) * 4;
    long stride = (long)gridDim.x * blockDim.x * 4;
    for (; i < n; i += stride) {
        float4 v = *(const float4*)(src + i);
        ushort4 o;
        o.x = f2bf(v.x); o.y = f2bf(v.y); o.z = f2bf(v.z); o.w = f2bf(v.w);
        *(ushort4*)(dst + i) = o;
    }
}

// ---------------- router: logits + softmax + top-2 ----------------
__global__ __launch_bounds__(256) void gate_kernel(
        const float* __restrict__ x, const float* __restrict__ gw,
        int* __restrict__ tke, float* __restrict__ tkp) {
    const int wave = threadIdx.x >> 5, lane = threadIdx.x & 31;
    const int tok = blockIdx.x * 8 + wave;
    const float* xr = x + (size_t)tok * HID;
    float acc[NEXP];
#pragma unroll
    for (int e = 0; e < NEXP; ++e) acc[e] = 0.f;
    for (int base = lane * 4; base < HID; base += 128) {
        float4 xv = *(const float4*)(xr + base);
#pragma unroll
        for (int e = 0; e < NEXP; ++e) {
            float4 gv = *(const float4*)(gw + e * HID + base);
            acc[e] = fmaf(xv.x, gv.x, acc[e]);
            acc[e] = fmaf(xv.y, gv.y, acc[e]);
            acc[e] = fmaf(xv.z, gv.z, acc[e]);
            acc[e] = fmaf(xv.w, gv.w, acc[e]);
        }
    }
#pragma unroll
    for (int off = 16; off > 0; off >>= 1)
#pragma unroll
        for (int e = 0; e < NEXP; ++e)
            acc[e] += __shfl_xor(acc[e], off, 32);
    if (lane == 0) {
        float mx = acc[0];
        for (int e = 1; e < NEXP; ++e) mx = fmaxf(mx, acc[e]);
        float p[NEXP], sum = 0.f;
        for (int e = 0; e < NEXP; ++e) { p[e] = __expf(acc[e] - mx); sum += p[e]; }
        float inv = 1.f / sum;
        int i0 = 0;
        for (int e = 1; e < NEXP; ++e) if (acc[e] > acc[i0]) i0 = e;
        int i1 = (i0 == 0) ? 1 : 0;
        for (int e = 0; e < NEXP; ++e) if (e != i0 && acc[e] > acc[i1]) i1 = e;
        tke[tok * 2 + 0] = i0; tke[tok * 2 + 1] = i1;
        tkp[tok * 2 + 0] = p[i0] * inv; tkp[tok * 2 + 1] = p[i1] * inv;
    }
}

__global__ void fill_neg1(int* __restrict__ p, int n) {
    int i = blockIdx.x * blockDim.x + threadIdx.x;
    if (i < n) p[i] = -1;
}

// ---------------- capacity assignment (ordered cumsum via ballot) ----------------
__global__ void assign_kernel(const int* __restrict__ tke,
                              int* __restrict__ slot_of,
                              int* __restrict__ dtok) {
    const int b = blockIdx.x;
    const int lane = threadIdx.x;     // blockDim = 32 (single wave32)
    int count[NEXP];
#pragma unroll
    for (int e = 0; e < NEXP; ++e) count[e] = 0;
    const unsigned below = (1u << lane) - 1u;
    for (int base = 0; base < SEQ * TOPK; base += 32) {
        const int i = base + lane;
        const int s = i >> 1, k = i & 1;
        const int idx = (b * SEQ + s) * TOPK + k;
        const int e = tke[idx];
        int slot = -1;
#pragma unroll
        for (int ex = 0; ex < NEXP; ++ex) {
            unsigned m = (unsigned)__ballot(e == ex);
            if (e == ex) {
                int pos = count[ex] + __popc(m & below);
                slot = (pos < CAP) ? pos : -1;
            }
            count[ex] += __popc(m);
        }
        slot_of[idx] = slot;
        if (slot >= 0) dtok[((size_t)e * BATCH + b) * CAP + slot] = s;
    }
}

// ---------------- gather tokens into capacity buffers (bf16) ----------------
__global__ __launch_bounds__(256) void dispatch_kernel(
        const float* __restrict__ x, const int* __restrict__ dtok,
        unsigned short* __restrict__ disp) {
    const int wave = threadIdx.x >> 5, lane = threadIdx.x & 31;
    const int row = blockIdx.x * 8 + wave;      // < ROWS
    const int b = (row / CAP) % BATCH;
    const int s = dtok[row];
    unsigned short* dst = disp + (size_t)row * HID;
    if (s >= 0) {
        const float* src = x + ((size_t)b * SEQ + s) * HID;
        for (int base = lane * 4; base < HID; base += 128) {
            float4 v = *(const float4*)(src + base);
            ushort4 o;
            o.x = f2bf(v.x); o.y = f2bf(v.y); o.z = f2bf(v.z); o.w = f2bf(v.w);
            *(ushort4*)(dst + base) = o;
        }
    } else {
        ushort4 z; z.x = z.y = z.z = z.w = 0;
        for (int base = lane * 4; base < HID; base += 128)
            *(ushort4*)(dst + base) = z;
    }
}

// ---------------- GEMM1: act = silu(disp@w1) * (disp@w3), bf16 out ----------------
__global__ __launch_bounds__(256) void gemm1_kernel(
        const unsigned short* __restrict__ disp,
        const unsigned short* __restrict__ w1b,
        const unsigned short* __restrict__ w3b,
        unsigned short* __restrict__ act) {
    constexpr int TM = 128, TN = 128, TK = 32, LDK = TK + 8;
    __shared__ unsigned short As [TM][LDK];
    __shared__ unsigned short Bs1[TN][LDK];
    __shared__ unsigned short Bs3[TN][LDK];

    const int eb = blockIdx.z;
    const int e  = eb / BATCH;
    const long arow0 = (long)eb * CAP + (long)blockIdx.y * TM;
    const unsigned short* Ag  = disp + arow0 * HID;
    const unsigned short* W1g = w1b + (size_t)e * HID * FFN + (size_t)blockIdx.x * TN;
    const unsigned short* W3g = w3b + (size_t)e * HID * FFN + (size_t)blockIdx.x * TN;

    const int tid = threadIdx.x, lane = tid & 31, wave = tid >> 5;
    const int wm = wave >> 1, wn = wave & 1;   // 4 x 2 wave grid
    const int hf = lane >> 4, lr = lane & 15;
    const unsigned as_off = (unsigned)(uintptr_t)(&As[0][0]);

    const floatx8 zero = {0.f, 0.f, 0.f, 0.f, 0.f, 0.f, 0.f, 0.f};
    floatx8 c1[2][4], c3[2][4];
#pragma unroll
    for (int mt = 0; mt < 2; ++mt)
#pragma unroll
        for (int nt = 0; nt < 4; ++nt) { c1[mt][nt] = zero; c3[mt][nt] = zero; }

    for (int kt = 0; kt < HID; kt += TK) {
        // A tile 128x32 via Tensor Data Mover (wave 0 issues, TENSORcnt tracked)
        if (wave == 0)
            tdm_load_tile_bf16(as_off, Ag + kt, TK, TM, HID);
        if (kt + TK < HID) {   // speculative prefetch of next weight tile
            __builtin_prefetch(W1g + (size_t)(kt + TK) * FFN + lane * 4, 0, 3);
            __builtin_prefetch(W3g + (size_t)(kt + TK) * FFN + lane * 4, 0, 3);
        }
        // B tiles 32x128, transposed into [n][k] so lanes read contiguous K
        for (int v = tid; v < (TK * TN) / 8; v += 256) {
            int k = v >> 4, n = (v & 15) << 3;
            uint4 d1 = *(const uint4*)(W1g + (size_t)(kt + k) * FFN + n);
            uint4 d3 = *(const uint4*)(W3g + (size_t)(kt + k) * FFN + n);
            unsigned short t1[8], t3[8];
            *(uint4*)t1 = d1; *(uint4*)t3 = d3;
#pragma unroll
            for (int j = 0; j < 8; ++j) { Bs1[n + j][k] = t1[j]; Bs3[n + j][k] = t3[j]; }
        }
        if (wave == 0) __builtin_amdgcn_s_wait_tensorcnt(0);
        __syncthreads();

        Frag a[2], b1f[4], b3f[4];
#pragma unroll
        for (int mt = 0; mt < 2; ++mt) {
            const unsigned short* pa = &As[wm * 32 + mt * 16 + lr][hf * 8];
            a[mt].q[0] = *(const uint4*)pa;          // K = 8*hf .. +7
            a[mt].q[1] = *(const uint4*)(pa + 16);   // K = 16+8*hf .. +7
        }
#pragma unroll
        for (int nt = 0; nt < 4; ++nt) {
            const unsigned short* p1 = &Bs1[wn * 64 + nt * 16 + lr][hf * 8];
            b1f[nt].q[0] = *(const uint4*)p1;        // K = 8*hf .. +7
            b1f[nt].q[1] = *(const uint4*)(p1 + 16); // K = 16+8*hf .. +7
            const unsigned short* p3 = &Bs3[wn * 64 + nt * 16 + lr][hf * 8];
            b3f[nt].q[0] = *(const uint4*)p3;
            b3f[nt].q[1] = *(const uint4*)(p3 + 16);
        }
#pragma unroll
        for (int mt = 0; mt < 2; ++mt)
#pragma unroll
            for (int nt = 0; nt < 4; ++nt) {
                c1[mt][nt] = __builtin_amdgcn_wmma_f32_16x16x32_bf16(
                    false, a[mt].v, false, b1f[nt].v, (short)0, c1[mt][nt], false, false);
                c3[mt][nt] = __builtin_amdgcn_wmma_f32_16x16x32_bf16(
                    false, a[mt].v, false, b3f[nt].v, (short)0, c3[mt][nt], false, false);
            }
        __syncthreads();
    }

    const size_t ncol0 = (size_t)blockIdx.x * TN + wn * 64;
#pragma unroll
    for (int mt = 0; mt < 2; ++mt)
#pragma unroll
        for (int nt = 0; nt < 4; ++nt)
#pragma unroll
            for (int r = 0; r < 8; ++r) {
                int row = wm * 32 + mt * 16 + hf * 8 + r;   // C: M = r + 8*half
                int col = nt * 16 + lr;                     //    N = lane%16
                float h1 = c1[mt][nt][r];
                float h3 = c3[mt][nt][r];
                float sg = 1.f / (1.f + __expf(-h1));
                act[(arow0 + row) * FFN + ncol0 + col] = f2bf(h1 * sg * h3);
            }
}

// ---------------- GEMM2: eout = act @ w2, f32 out ----------------
__global__ __launch_bounds__(256) void gemm2_kernel(
        const unsigned short* __restrict__ act,
        const unsigned short* __restrict__ w2b,
        float* __restrict__ eout) {
    constexpr int TM = 128, TN = 128, TK = 32, LDK = TK + 8;
    __shared__ unsigned short As[TM][LDK];
    __shared__ unsigned short Bs[TN][LDK];

    const int eb = blockIdx.z;
    const int e  = eb / BATCH;
    const long arow0 = (long)eb * CAP + (long)blockIdx.y * TM;
    const unsigned short* Ag = act + arow0 * FFN;
    const unsigned short* Wg = w2b + (size_t)e * FFN * HID + (size_t)blockIdx.x * TN;

    const int tid = threadIdx.x, lane = tid & 31, wave = tid >> 5;
    const int wm = wave >> 1, wn = wave & 1;
    const int hf = lane >> 4, lr = lane & 15;
    const unsigned as_off = (unsigned)(uintptr_t)(&As[0][0]);

    const floatx8 zero = {0.f, 0.f, 0.f, 0.f, 0.f, 0.f, 0.f, 0.f};
    floatx8 c[2][4];
#pragma unroll
    for (int mt = 0; mt < 2; ++mt)
#pragma unroll
        for (int nt = 0; nt < 4; ++nt) c[mt][nt] = zero;

    for (int kt = 0; kt < FFN; kt += TK) {
        if (wave == 0)
            tdm_load_tile_bf16(as_off, Ag + kt, TK, TM, FFN);
        if (kt + TK < FFN)
            __builtin_prefetch(Wg + (size_t)(kt + TK) * HID + lane * 4, 0, 3);
        for (int v = tid; v < (TK * TN) / 8; v += 256) {
            int k = v >> 4, n = (v & 15) << 3;
            uint4 d = *(const uint4*)(Wg + (size_t)(kt + k) * HID + n);
            unsigned short t[8];
            *(uint4*)t = d;
#pragma unroll
            for (int j = 0; j < 8; ++j) Bs[n + j][k] = t[j];
        }
        if (wave == 0) __builtin_amdgcn_s_wait_tensorcnt(0);
        __syncthreads();

        Frag a[2], bf[4];
#pragma unroll
        for (int mt = 0; mt < 2; ++mt) {
            const unsigned short* pa = &As[wm * 32 + mt * 16 + lr][hf * 8];
            a[mt].q[0] = *(const uint4*)pa;
            a[mt].q[1] = *(const uint4*)(pa + 16);
        }
#pragma unroll
        for (int nt = 0; nt < 4; ++nt) {
            const unsigned short* pb = &Bs[wn * 64 + nt * 16 + lr][hf * 8];
            bf[nt].q[0] = *(const uint4*)pb;
            bf[nt].q[1] = *(const uint4*)(pb + 16);
        }
#pragma unroll
        for (int mt = 0; mt < 2; ++mt)
#pragma unroll
            for (int nt = 0; nt < 4; ++nt)
                c[mt][nt] = __builtin_amdgcn_wmma_f32_16x16x32_bf16(
                    false, a[mt].v, false, bf[nt].v, (short)0, c[mt][nt], false, false);
        __syncthreads();
    }

    const size_t ncol0 = (size_t)blockIdx.x * TN + wn * 64;
#pragma unroll
    for (int mt = 0; mt < 2; ++mt)
#pragma unroll
        for (int nt = 0; nt < 4; ++nt)
#pragma unroll
            for (int r = 0; r < 8; ++r) {
                int row = wm * 32 + mt * 16 + hf * 8 + r;
                int col = nt * 16 + lr;
                eout[(arow0 + row) * HID + ncol0 + col] = c[mt][nt][r];
            }
}

// ---------------- weighted combine back to token order ----------------
__global__ __launch_bounds__(256) void combine_kernel(
        const float* __restrict__ eout, const int* __restrict__ tke,
        const float* __restrict__ tkp, const int* __restrict__ slot_of,
        float* __restrict__ out) {
    const int wave = threadIdx.x >> 5, lane = threadIdx.x & 31;
    const int tok = blockIdx.x * 8 + wave;
    const int b = tok / SEQ;
    const int e0 = tke[tok * 2 + 0], e1 = tke[tok * 2 + 1];
    const float p0 = tkp[tok * 2 + 0], p1 = tkp[tok * 2 + 1];
    const int s0 = slot_of[tok * 2 + 0], s1 = slot_of[tok * 2 + 1];
    const float* r0 = (s0 >= 0) ? eout + (((size_t)e0 * BATCH + b) * CAP + s0) * HID : nullptr;
    const float* r1 = (s1 >= 0) ? eout + (((size_t)e1 * BATCH + b) * CAP + s1) * HID : nullptr;
    float* o = out + (size_t)tok * HID;
    for (int base = lane * 4; base < HID; base += 128) {
        float4 acc = {0.f, 0.f, 0.f, 0.f};
        if (r0) {
            float4 v = *(const float4*)(r0 + base);
            acc.x += p0 * v.x; acc.y += p0 * v.y; acc.z += p0 * v.z; acc.w += p0 * v.w;
        }
        if (r1) {
            float4 v = *(const float4*)(r1 + base);
            acc.x += p1 * v.x; acc.y += p1 * v.y; acc.z += p1 * v.z; acc.w += p1 * v.w;
        }
        *(float4*)(o + base) = acc;
    }
}

// ---------------- host launcher ----------------
extern "C" void kernel_launch(void* const* d_in, const int* in_sizes, int n_in,
                              void* d_out, int out_size, void* d_ws, size_t ws_size,
                              hipStream_t stream) {
    (void)in_sizes; (void)n_in; (void)out_size; (void)ws_size;
    const float* hs = (const float*)d_in[0];   // hidden_states [B,S,H]
    const float* gw = (const float*)d_in[1];   // gate_w [E,H]
    const float* w1 = (const float*)d_in[2];   // w1 [E,H,F]
    const float* w2 = (const float*)d_in[3];   // w2 [E,F,H]
    const float* w3 = (const float*)d_in[4];   // w3 [E,H,F]

    char* ws = (char*)d_ws;
    unsigned short* w1b = (unsigned short*)(ws + OFF_W1);
    unsigned short* w3b = (unsigned short*)(ws + OFF_W3);
    unsigned short* w2b = (unsigned short*)(ws + OFF_W2);
    unsigned short* dsp = (unsigned short*)(ws + OFF_DSP);
    unsigned short* act = (unsigned short*)(ws + OFF_ACT);
    float* eo  = (float*)(ws + OFF_EO);
    int*   tke = (int*)  (ws + OFF_TE);
    float* tkp = (float*)(ws + OFF_TP);
    int*   slt = (int*)  (ws + OFF_SL);
    int*   dtk = (int*)  (ws + OFF_DT);

    cvt_bf16_kernel<<<4096, 256, 0, stream>>>(w1, w1b, (long)NW);
    cvt_bf16_kernel<<<4096, 256, 0, stream>>>(w3, w3b, (long)NW);
    cvt_bf16_kernel<<<4096, 256, 0, stream>>>(w2, w2b, (long)NW);

    gate_kernel<<<TOKENS / 8, 256, 0, stream>>>(hs, gw, tke, tkp);
    fill_neg1<<<(ROWS + 255) / 256, 256, 0, stream>>>(dtk, ROWS);
    assign_kernel<<<BATCH, 32, 0, stream>>>(tke, slt, dtk);
    dispatch_kernel<<<ROWS / 8, 256, 0, stream>>>(hs, dtk, dsp);

    gemm1_kernel<<<dim3(FFN / 128, CAP / 128, NEXP * BATCH), 256, 0, stream>>>(dsp, w1b, w3b, act);
    gemm2_kernel<<<dim3(HID / 128, CAP / 128, NEXP * BATCH), 256, 0, stream>>>(act, w2b, eo);

    combine_kernel<<<TOKENS / 8, 256, 0, stream>>>(eo, tke, tkp, slt, (float*)d_out);
}
